// GCN_86870008529232
// MI455X (gfx1250) — matrix operand order repaired and checked by reading, not compiled, not run on hardware
//
#include <hip/hip_runtime.h>
#include <hip/hip_bf16.h>
#include <cstdint>

#define BB 8
#define NN 2048
#define DD 256

typedef __attribute__((ext_vector_type(16))) __bf16 v16bf;
typedef __attribute__((ext_vector_type(8)))  float  v8f;

union FragU { uint4 u[2]; v16bf v; };
union Pack8 { unsigned short s[8]; uint4 u; };

__device__ __forceinline__ unsigned short f2bf(float f) {
  unsigned int x = __float_as_uint(f);
  unsigned int r = (x + 0x7FFFu + ((x >> 16) & 1u)) >> 16;
  return (unsigned short)r;
}

// A-fragment: 16x32 bf16 tile, rows i = it..it+15, k = k0..k0+31.
// Lanes 0-15 hold K {0..7,16..23}; lanes 16-31 hold K {8..15,24..31}. (ISA 7.12.2)
__device__ __forceinline__ v16bf load_row_frag(const unsigned short* base, int stride,
                                               int it, int k0, int lane) {
  int row = it + (lane & 15);
  int ko  = k0 + ((lane & 16) >> 1);          // +8 for upper half-wave
  const unsigned short* p = base + (size_t)row * stride + ko;
  FragU f;
  f.u[0] = *(const uint4*)(p);                // K ko..ko+7
  f.u[1] = *(const uint4*)(p + 16);           // K ko+16..ko+23
  return f.v;
}

// B-fragment: 32x16 bf16; column j of B = row j of R (row-major over K).
// Lanes 0-15 hold K {0..7,16..23}; lanes 16-31 hold K {8..15,24..31}.
__device__ __forceinline__ v16bf load_col_frag(const unsigned short* base, int stride,
                                               int jt, int k0, int lane) {
  int row = jt + (lane & 15);
  int ko  = k0 + ((lane & 16) >> 1);          // +8 for upper half-wave
  const unsigned short* p = base + (size_t)row * stride + ko;
  FragU f;
  f.u[0] = *(const uint4*)(p);
  f.u[1] = *(const uint4*)(p + 16);
  return f.v;
}

// Fragment set for a 32x64 wave tile k-step: 2 A-frags + 4 B-frags.
struct Frags {
  v16bf a0, a1, b0, b1, b2, b3;
};

__device__ __forceinline__ Frags load_frags(const unsigned short* L, int ls,
                                            const unsigned short* R, int rs,
                                            int it, int jt, int k0, int lane) {
  Frags f;
  f.a0 = load_row_frag(L, ls, it,      k0, lane);
  f.a1 = load_row_frag(L, ls, it + 16, k0, lane);
  f.b0 = load_col_frag(R, rs, jt,      k0, lane);
  f.b1 = load_col_frag(R, rs, jt + 16, k0, lane);
  f.b2 = load_col_frag(R, rs, jt + 32, k0, lane);
  f.b3 = load_col_frag(R, rs, jt + 48, k0, lane);
  return f;
}

__device__ __forceinline__ void mma8(v8f acc[2][4], const Frags& f) {
  acc[0][0] = __builtin_amdgcn_wmma_f32_16x16x32_bf16(false, f.a0, false, f.b0, (short)0, acc[0][0], false, false);
  acc[1][0] = __builtin_amdgcn_wmma_f32_16x16x32_bf16(false, f.a1, false, f.b0, (short)0, acc[1][0], false, false);
  acc[0][1] = __builtin_amdgcn_wmma_f32_16x16x32_bf16(false, f.a0, false, f.b1, (short)0, acc[0][1], false, false);
  acc[1][1] = __builtin_amdgcn_wmma_f32_16x16x32_bf16(false, f.a1, false, f.b1, (short)0, acc[1][1], false, false);
  acc[0][2] = __builtin_amdgcn_wmma_f32_16x16x32_bf16(false, f.a0, false, f.b2, (short)0, acc[0][2], false, false);
  acc[1][2] = __builtin_amdgcn_wmma_f32_16x16x32_bf16(false, f.a1, false, f.b2, (short)0, acc[1][2], false, false);
  acc[0][3] = __builtin_amdgcn_wmma_f32_16x16x32_bf16(false, f.a0, false, f.b3, (short)0, acc[0][3], false, false);
  acc[1][3] = __builtin_amdgcn_wmma_f32_16x16x32_bf16(false, f.a1, false, f.b3, (short)0, acc[1][3], false, false);
}

// ---------- prep: weights fp32->bf16, denom := 1.0 ----------
__global__ void k_prep(const float* __restrict__ Ww, const float* __restrict__ Lw,
                       float* __restrict__ denom,
                       unsigned short* __restrict__ Wwb, unsigned short* __restrict__ Lwb) {
  int i = blockIdx.x * 256 + threadIdx.x;
  if (i < DD * DD)      Wwb[i] = f2bf(Ww[i]);
  if (i < 2 * DD * DD)  Lwb[i] = f2bf(Lw[i]);
  if (i < BB * NN)      denom[i] = 1.0f;
}

// ---------- transpose+convert: x fp32 [B][NN][DD] -> bf16 row-major + bf16 [B][DD][NN] ----------
__global__ void k_trans(const float* __restrict__ x, unsigned short* __restrict__ xrow,
                        unsigned short* __restrict__ xcol) {
  __shared__ unsigned short tile[32][33];
  int b = blockIdx.z;
  int d0 = blockIdx.x * 32, n0 = blockIdx.y * 32;
  int tx = threadIdx.x, ty = threadIdx.y;
  #pragma unroll
  for (int i = 0; i < 32; i += 8) {
    int n = n0 + ty + i, d = d0 + tx;
    unsigned short v = f2bf(x[((size_t)b * NN + n) * DD + d]);
    xrow[((size_t)b * NN + n) * DD + d] = v;
    tile[ty + i][tx] = v;
  }
  __syncthreads();
  #pragma unroll
  for (int i = 0; i < 32; i += 8) {
    int d = d0 + ty + i, n = n0 + tx;
    xcol[((size_t)b * DD + d) * NN + n] = tile[tx][ty + i];
  }
}

// ---------- q[n][e] = x[n]·Ww[e] + bias[e] ; L = Ww rows e, R = x rows n ----------
__global__ void k_q(const unsigned short* __restrict__ Wwb, const unsigned short* __restrict__ xb,
                    const float* __restrict__ bias, unsigned short* __restrict__ qb) {
  int lane = threadIdx.x & 31, wave = threadIdx.x >> 5;
  int b = blockIdx.z;
  int it = blockIdx.y * 64 + wave * 16;       // e
  int jt = blockIdx.x * 64;                   // n
  const unsigned short* xbB = xb + (size_t)b * NN * DD;
  v8f acc[4] = {};
  for (int k0 = 0; k0 < DD; k0 += 32) {
    v16bf a = load_row_frag(Wwb, DD, it, k0, lane);
    #pragma unroll
    for (int t = 0; t < 4; ++t) {
      v16bf bf = load_col_frag(xbB, DD, jt + t * 16, k0, lane);
      acc[t] = __builtin_amdgcn_wmma_f32_16x16x32_bf16(false, a, false, bf,
                                                       (short)0, acc[t], false, false);
    }
  }
  int e0 = it + ((lane & 16) >> 1);           // rows i = e0..e0+7 per lane
  float b8[8];
  #pragma unroll
  for (int r = 0; r < 8; ++r) b8[r] = bias[e0 + r];
  #pragma unroll
  for (int t = 0; t < 4; ++t) {
    int n = jt + t * 16 + (lane & 15);
    Pack8 o;
    #pragma unroll
    for (int r = 0; r < 8; ++r) o.s[r] = f2bf(acc[t][r] + b8[r]);
    *(uint4*)(qb + ((size_t)b * NN + n) * DD + e0) = o.u;
  }
}

// ---------- combined[n][m] = adj[n][m]*exp(x[m]·q[n]); denom[n] += rowsum ----------
// 32x64 wave tile, double-buffered fragment loads (8 WMMAs per k-step).
__global__ void k_s(const unsigned short* __restrict__ xb, const unsigned short* __restrict__ qb,
                    const float* __restrict__ adj, unsigned short* __restrict__ comb,
                    float* __restrict__ denom) {
  int lane = threadIdx.x & 31, wave = threadIdx.x >> 5;
  int b = blockIdx.z;
  int it = blockIdx.y * 128 + wave * 32;      // m (32 rows per wave)
  int jt = blockIdx.x * 64;                   // n
  const unsigned short* xbB = xb + (size_t)b * NN * DD;
  const unsigned short* qbB = qb + (size_t)b * NN * DD;
  v8f acc[2][4] = {};
  Frags cur = load_frags(xbB, DD, qbB, DD, it, jt, 0, lane);
  for (int k0 = 0; k0 < DD - 32; k0 += 32) {
    Frags nxt = load_frags(xbB, DD, qbB, DD, it, jt, k0 + 32, lane);
    mma8(acc, cur);
    cur = nxt;
  }
  mma8(acc, cur);

  #pragma unroll
  for (int mi = 0; mi < 2; ++mi) {
    int m0 = it + mi * 16 + ((lane & 16) >> 1);
    #pragma unroll
    for (int t = 0; t < 4; ++t) {
      int n = jt + t * 16 + (lane & 15);
      const float* ap = adj + ((size_t)b * NN + n) * NN + m0;
      float4 a0 = *(const float4*)(ap);
      float4 a1 = *(const float4*)(ap + 4);
      float av[8] = {a0.x, a0.y, a0.z, a0.w, a1.x, a1.y, a1.z, a1.w};
      float rowsum = 0.0f;
      Pack8 o;
      #pragma unroll
      for (int r = 0; r < 8; ++r) {
        float v = av[r] * __expf(acc[mi][t][r]);
        rowsum += v;
        o.s[r] = f2bf(v);
      }
      *(uint4*)(comb + ((size_t)b * NN + n) * NN + m0) = o.u;
      rowsum += __shfl_xor(rowsum, 16);       // combine the two m-halves of this n
      if (lane < 16) atomicAdd(&denom[b * NN + n], rowsum);
    }
  }
}

// ---------- h[n][d] = Ax[n][d]/denom[n] + x[n][d] ; L = xT rows d, R = comb rows n ----------
// 32x64 wave tile, K = NN, double-buffered fragment loads (512 WMMAs/wave).
__global__ void k_ax(const unsigned short* __restrict__ xT, const unsigned short* __restrict__ comb,
                     const float* __restrict__ denom, const float* __restrict__ xf,
                     unsigned short* __restrict__ hb) {
  int lane = threadIdx.x & 31, wave = threadIdx.x >> 5;
  int b = blockIdx.z;
  int it = blockIdx.y * 128 + wave * 32;      // d (32 rows per wave)
  int jt = blockIdx.x * 64;                   // n
  const unsigned short* xTB = xT + (size_t)b * DD * NN;
  const unsigned short* cB  = comb + (size_t)b * NN * NN;
  v8f acc[2][4] = {};
  Frags cur = load_frags(xTB, NN, cB, NN, it, jt, 0, lane);
  for (int k0 = 0; k0 < NN - 32; k0 += 32) {
    Frags nxt = load_frags(xTB, NN, cB, NN, it, jt, k0 + 32, lane);
    mma8(acc, cur);
    cur = nxt;
  }
  mma8(acc, cur);

  #pragma unroll
  for (int t = 0; t < 4; ++t) {
    int n = jt + t * 16 + (lane & 15);
    float s = 1.0f / denom[b * NN + n];       // denom already includes +1
    #pragma unroll
    for (int mi = 0; mi < 2; ++mi) {
      int d0 = it + mi * 16 + ((lane & 16) >> 1);
      const float* xp = xf + ((size_t)b * NN + n) * DD + d0;
      float4 x0 = *(const float4*)(xp);
      float4 x1 = *(const float4*)(xp + 4);
      float xv[8] = {x0.x, x0.y, x0.z, x0.w, x1.x, x1.y, x1.z, x1.w};
      Pack8 o;
      #pragma unroll
      for (int r = 0; r < 8; ++r) o.s[r] = f2bf(acc[mi][t][r] * s + xv[r]);
      *(uint4*)(hb + ((size_t)b * NN + n) * DD + d0) = o.u;
    }
  }
}

// ---------- out[n][e] = relu(h[n]·W[e] + 2*bias[e]) ; L = W rows e, R = h rows n ----------
__global__ void k_lin(const unsigned short* __restrict__ Wb, const unsigned short* __restrict__ hb,
                      const float* __restrict__ bias, float* __restrict__ out) {
  int lane = threadIdx.x & 31, wave = threadIdx.x >> 5;
  int b = blockIdx.z;
  int it = blockIdx.y * 64 + wave * 16;       // e
  int jt = blockIdx.x * 64;                   // n
  const unsigned short* hB = hb + (size_t)b * NN * DD;
  v8f acc[4] = {};
  for (int k0 = 0; k0 < DD; k0 += 32) {
    v16bf a = load_row_frag(Wb, DD, it, k0, lane);
    #pragma unroll
    for (int t = 0; t < 4; ++t) {
      v16bf bf = load_col_frag(hB, DD, jt + t * 16, k0, lane);
      acc[t] = __builtin_amdgcn_wmma_f32_16x16x32_bf16(false, a, false, bf,
                                                       (short)0, acc[t], false, false);
    }
  }
  int e0 = it + ((lane & 16) >> 1);
  const float* bp = bias + e0;
  float4 b0 = *(const float4*)(bp);
  float4 b1 = *(const float4*)(bp + 4);
  float bv[8] = {b0.x, b0.y, b0.z, b0.w, b1.x, b1.y, b1.z, b1.w};
  #pragma unroll
  for (int t = 0; t < 4; ++t) {
    int n = jt + t * 16 + (lane & 15);
    float* op = out + ((size_t)b * NN + n) * DD + e0;
    float v[8];
    #pragma unroll
    for (int r = 0; r < 8; ++r) {
      float y = acc[t][r] + 2.0f * bv[r];
      v[r] = y > 0.0f ? y : 0.0f;
    }
    *(float4*)(op)     = make_float4(v[0], v[1], v[2], v[3]);
    *(float4*)(op + 4) = make_float4(v[4], v[5], v[6], v[7]);
  }
}

extern "C" void kernel_launch(void* const* d_in, const int* in_sizes, int n_in,
                              void* d_out, int out_size, void* d_ws, size_t ws_size,
                              hipStream_t stream) {
  const float* x0  = (const float*)d_in[0];   // [B,N,D]
  const float* adj = (const float*)d_in[1];   // [B,N,N]
  const float* Ww  = (const float*)d_in[2];   // [D,D]
  const float* Wb_ = (const float*)d_in[3];   // [D]
  const float* Lw  = (const float*)d_in[4];   // [L,D,D]
  const float* Lb  = (const float*)d_in[5];   // [L,D]
  float* out = (float*)d_out;

  char* p = (char*)d_ws;
  auto take = [&](size_t bytes) -> void* {
    void* r = (void*)p;
    p += (bytes + 255) & ~(size_t)255;
    return r;
  };
  unsigned short* x0b  = (unsigned short*)take((size_t)BB * NN * DD * 2);
  unsigned short* x0T  = (unsigned short*)take((size_t)BB * DD * NN * 2);
  unsigned short* qb   = (unsigned short*)take((size_t)BB * NN * DD * 2);
  unsigned short* comb = (unsigned short*)take((size_t)BB * NN * NN * 2);
  float*          den  = (float*)take((size_t)BB * NN * 4);
  unsigned short* hb   = (unsigned short*)take((size_t)BB * NN * DD * 2);
  float*          x1f  = (float*)take((size_t)BB * NN * DD * 4);
  unsigned short* x1T  = (unsigned short*)take((size_t)BB * DD * NN * 2);
  unsigned short* Wwb  = (unsigned short*)take((size_t)DD * DD * 2);
  unsigned short* Lwb  = (unsigned short*)take((size_t)2 * DD * DD * 2);

  dim3 blk128(128);
  dim3 gQ (NN / 64, DD / 64, BB);             // 16-row wave tiles over [D x N]
  dim3 gS (NN / 64, NN / 128, BB);            // 32-row wave tiles over [N x N]
  dim3 gAX(NN / 64, DD / 128, BB);            // 32-row wave tiles over [D x N]
  dim3 trGrid(DD / 32, NN / 32, BB), trBlk(32, 8);

  k_prep <<<dim3((2 * DD * DD + 255) / 256), dim3(256), 0, stream>>>(Ww, Lw, den, Wwb, Lwb);
  k_trans<<<trGrid, trBlk, 0, stream>>>(x0, x0b, x0T);
  k_q    <<<gQ, blk128, 0, stream>>>(Wwb, x0b, Wb_, qb);
  k_s    <<<gS, blk128, 0, stream>>>(x0b, qb, adj, comb, den);
  // layer 0
  k_ax   <<<gAX, blk128, 0, stream>>>(x0T, comb, den, x0, hb);
  k_lin  <<<gQ, blk128, 0, stream>>>(Lwb, hb, Lb, x1f);
  // layer 1
  k_trans<<<trGrid, trBlk, 0, stream>>>(x1f, qb /*dead, reused as scratch*/, x1T);
  k_ax   <<<gAX, blk128, 0, stream>>>(x1T, comb, den, x1f, hb);
  k_lin  <<<gQ, blk128, 0, stream>>>(Lwb + (size_t)DD * DD, hb, Lb + DD, out);
}
